// GNN_62182536511521
// MI455X (gfx1250) — matrix operand order repaired and checked by reading, not compiled, not run on hardware
//
#include <hip/hip_runtime.h>
#include <math.h>

typedef __attribute__((ext_vector_type(16))) __bf16 v16bf;
typedef __attribute__((ext_vector_type(8)))  float  v8f;

#define NN    10000
#define NH1   32
#define NH2   16
#define MT    625           // NN/16 row tiles
#define KT1   313           // ceil(NN/32) k-tiles for GEMM1
#define NCHK  20            // split-K chunks for GEMM1
#define KPC   16            // k-tiles per chunk (16*20 >= 313)

static __device__ __forceinline__ v8f wmma_bf16(v16bf a, v16bf b, v8f c) {
  // (neg_a, A, neg_b, B, c_mod, C, reuse_a, reuse_b)
  return __builtin_amdgcn_wmma_f32_16x16x32_bf16(false, a, false, b, (short)0, c,
                                                 false, false);
}

// ---------------- utility kernels ----------------

__global__ void fill_f32(float* __restrict__ p, float v, int n) {
  int i = blockIdx.x * blockDim.x + threadIdx.x;
  if (i < n) p[i] = v;
}

__global__ void deg_edges(const int* __restrict__ dst, float* __restrict__ deg, int E) {
  int e = blockIdx.x * blockDim.x + threadIdx.x;
  if (e < E) atomicAdd(&deg[dst[e]], 1.0f);
}

__global__ void rsqrt_f32(const float* __restrict__ deg, float* __restrict__ dinv, int n) {
  int i = blockIdx.x * blockDim.x + threadIdx.x;
  if (i < n) dinv[i] = rsqrtf(deg[i]);
}

// out[i,f] = b[f] + dinv[i]^2 * h[i,f]   (self-loop term + bias), F = 1<<shift
__global__ void self_bias(const float* __restrict__ h, const float* __restrict__ dinv,
                          const float* __restrict__ b, float* __restrict__ out,
                          int total, int shift) {
  int tid = blockIdx.x * blockDim.x + threadIdx.x;
  if (tid >= total) return;
  int i = tid >> shift;
  int f = tid & ((1 << shift) - 1);
  float w = dinv[i];
  out[tid] = b[f] + w * w * h[tid];
}

// Pre-swizzle a [K, ncols] fp32 weight matrix into per-lane WMMA B-operand
// bf16 hi/lo planes. Layout per 16x16(n) x 32(k) tile t = kt*ntiles + nt:
//   lane L holds column nt*16 + (L&15), K = kt*32 + (i>>3)*16 + (L>>4)*8 + (i&7), i = 0..15
__global__ void prep_bswz(const float* __restrict__ W, int K, int ncols,
                          int ktiles, int ntiles,
                          __bf16* __restrict__ hi, __bf16* __restrict__ lo) {
  int tid = blockIdx.x * blockDim.x + threadIdx.x;
  int total = ktiles * ntiles * 512;
  if (tid >= total) return;
  int i    = tid & 15;
  int lane = (tid >> 4) & 31;
  int t    = tid >> 9;
  int nt   = t % ntiles;
  int kt   = t / ntiles;
  int col  = nt * 16 + (lane & 15);
  int k    = kt * 32 + ((i >> 3) << 4) + ((lane >> 4) << 3) + (i & 7);
  float v  = (k < K) ? W[(size_t)k * ncols + col] : 0.0f;
  __bf16 h = (__bf16)v;
  hi[tid]  = h;
  lo[tid]  = (__bf16)(v - (float)h);
}

// ---------------- GEMM1: h1 += x[10000,10000] @ W1[10000,32] (split-K) ----------------

__global__ void __launch_bounds__(256)
gemm1_wmma(const float* __restrict__ x,
           const __bf16* __restrict__ Bhi, const __bf16* __restrict__ Blo,
           float* __restrict__ h1) {
  int wid  = (blockIdx.x * blockDim.x + threadIdx.x) >> 5;
  int lane = threadIdx.x & 31;
  if (wid >= MT * NCHK) return;
  int m   = wid % MT;
  int c   = wid / MT;
  int kt0 = c * KPC;
  int kt1 = kt0 + KPC; if (kt1 > KT1) kt1 = KT1;

  int row = m * 16 + (lane & 15);
  const float* __restrict__ xrow = x + (size_t)row * NN;
  int koff = (lane >> 4) * 8;   // A layout: half-waves interleave K groups of 8

  v8f acc0 = {};
  v8f acc1 = {};

  for (int kt = kt0; kt < kt1; ++kt) {
    int k0 = kt * 32;
    __builtin_prefetch(xrow + k0 + 32 + koff, 0, 3);  // global_prefetch next tile
    float a[16];
    if (k0 + 32 <= NN) {
      float4 p0 = *(const float4*)(xrow + k0 + koff);
      float4 p1 = *(const float4*)(xrow + k0 + koff + 4);
      float4 p2 = *(const float4*)(xrow + k0 + 16 + koff);
      float4 p3 = *(const float4*)(xrow + k0 + 16 + koff + 4);
      a[0]=p0.x;  a[1]=p0.y;  a[2]=p0.z;  a[3]=p0.w;
      a[4]=p1.x;  a[5]=p1.y;  a[6]=p1.z;  a[7]=p1.w;
      a[8]=p2.x;  a[9]=p2.y;  a[10]=p2.z; a[11]=p2.w;
      a[12]=p3.x; a[13]=p3.y; a[14]=p3.z; a[15]=p3.w;
    } else {
#pragma unroll
      for (int i = 0; i < 8; ++i) {
        int ka = k0 + koff + i;
        int kb = k0 + 16 + koff + i;
        a[i]     = (ka < NN) ? xrow[ka] : 0.0f;
        a[8 + i] = (kb < NN) ? xrow[kb] : 0.0f;
      }
    }
    v16bf ah, al;
#pragma unroll
    for (int i = 0; i < 16; ++i) {
      __bf16 h = (__bf16)a[i];
      ah[i] = h;
      al[i] = (__bf16)(a[i] - (float)h);
    }
    const __bf16* __restrict__ bp = Bhi + ((size_t)kt * 2) * 512 + lane * 16;
    const __bf16* __restrict__ lp = Blo + ((size_t)kt * 2) * 512 + lane * 16;
    v16bf bh0 = *(const v16bf*)(bp);
    v16bf bh1 = *(const v16bf*)(bp + 512);
    v16bf bl0 = *(const v16bf*)(lp);
    v16bf bl1 = *(const v16bf*)(lp + 512);

    acc0 = wmma_bf16(ah, bh0, acc0);
    acc1 = wmma_bf16(ah, bh1, acc1);
    acc0 = wmma_bf16(al, bh0, acc0);
    acc1 = wmma_bf16(al, bh1, acc1);
    acc0 = wmma_bf16(ah, bl0, acc0);
    acc1 = wmma_bf16(ah, bl1, acc1);
  }

  // D layout: VGPR v, lanes 0-15 -> M=v, lanes 16-31 -> M=8+v ; N = lane&15
  int mrow = m * 16 + (lane >> 4) * 8;
  int n = lane & 15;
#pragma unroll
  for (int v = 0; v < 8; ++v) {
    atomicAdd(&h1[(size_t)(mrow + v) * NH1 + n],      acc0[v]);
    atomicAdd(&h1[(size_t)(mrow + v) * NH1 + 16 + n], acc1[v]);
  }
}

// ---------------- GEMM2: h2 = relu(out1)[10000,32] @ W2[32,16] ----------------

__global__ void __launch_bounds__(256)
gemm2_wmma(const float* __restrict__ out1,
           const __bf16* __restrict__ Bhi, const __bf16* __restrict__ Blo,
           float* __restrict__ h2) {
  int wid  = (blockIdx.x * blockDim.x + threadIdx.x) >> 5;
  int lane = threadIdx.x & 31;
  if (wid >= MT) return;
  int row  = wid * 16 + (lane & 15);
  int koff = (lane >> 4) * 8;

  float a[16];
#pragma unroll
  for (int i = 0; i < 8; ++i) {
    float v0 = out1[(size_t)row * NH1 + koff + i];
    float v1 = out1[(size_t)row * NH1 + 16 + koff + i];
    a[i]     = v0 > 0.0f ? v0 : 0.0f;   // fused ReLU
    a[8 + i] = v1 > 0.0f ? v1 : 0.0f;
  }
  v16bf ah, al;
#pragma unroll
  for (int i = 0; i < 16; ++i) {
    __bf16 h = (__bf16)a[i];
    ah[i] = h;
    al[i] = (__bf16)(a[i] - (float)h);
  }
  v16bf bh = *(const v16bf*)(Bhi + lane * 16);
  v16bf bl = *(const v16bf*)(Blo + lane * 16);
  v8f acc = {};
  acc = wmma_bf16(ah, bh, acc);
  acc = wmma_bf16(al, bh, acc);
  acc = wmma_bf16(ah, bl, acc);

  int mrow = wid * 16 + (lane >> 4) * 8;
  int n = lane & 15;
#pragma unroll
  for (int v = 0; v < 8; ++v)
    h2[(size_t)(mrow + v) * NH2 + n] = acc[v];
}

// ---------------- edge aggregation (scatter-add) ----------------

__global__ void agg_edges32(const int* __restrict__ src, const int* __restrict__ dst,
                            const float* __restrict__ dinv, const float* __restrict__ h,
                            float* __restrict__ out, int E) {
  int tid = blockIdx.x * blockDim.x + threadIdx.x;
  int e = tid >> 5;
  if (e >= E) return;
  int f = tid & 31;
  int s = src[e], d = dst[e];
  float w = dinv[s] * dinv[d];
  atomicAdd(&out[(size_t)d * NH1 + f], w * h[(size_t)s * NH1 + f]);
}

__global__ void agg_edges16(const int* __restrict__ src, const int* __restrict__ dst,
                            const float* __restrict__ dinv, const float* __restrict__ h,
                            float* __restrict__ out, int E) {
  int tid = blockIdx.x * blockDim.x + threadIdx.x;
  int e = tid >> 4;
  if (e >= E) return;
  int f = tid & 15;
  int s = src[e], d = dst[e];
  float w = dinv[s] * dinv[d];
  atomicAdd(&out[(size_t)d * NH2 + f], w * h[(size_t)s * NH2 + f]);
}

// ---------------- pair scoring ----------------

__global__ void pair_score(const float* __restrict__ h2, const int* __restrict__ pair,
                           const float* __restrict__ Wfc, const float* __restrict__ bfc,
                           float* __restrict__ out, int P) {
  int p = blockIdx.x * blockDim.x + threadIdx.x;
  if (p >= P) return;
  int a = pair[2 * p];
  int b = pair[2 * p + 1];
  float z = bfc[0];
#pragma unroll
  for (int f = 0; f < NH2; ++f) z += h2[(size_t)a * NH2 + f] * Wfc[f];
#pragma unroll
  for (int f = 0; f < NH2; ++f) z += h2[(size_t)b * NH2 + f] * Wfc[NH2 + f];
  out[p] = 1.0f / (1.0f + expf(-z));
}

// ---------------- host launcher ----------------

extern "C" void kernel_launch(void* const* d_in, const int* in_sizes, int n_in,
                              void* d_out, int out_size, void* d_ws, size_t ws_size,
                              hipStream_t stream) {
  const float* x   = (const float*)d_in[0];
  const int*   ei  = (const int*)d_in[1];   // [2, E]
  const int*   pi  = (const int*)d_in[2];   // [P, 2]
  const float* W1  = (const float*)d_in[3];
  const float* b1  = (const float*)d_in[4];
  const float* W2  = (const float*)d_in[5];
  const float* b2  = (const float*)d_in[6];
  const float* Wfc = (const float*)d_in[7];
  const float* bfc = (const float*)d_in[8];
  float* out = (float*)d_out;

  const int E = in_sizes[1] / 2;
  const int P = in_sizes[2] / 2;
  const int* esrc = ei;
  const int* edst = ei + E;

  // carve workspace
  char* base = (char*)d_ws;
  size_t off = 0;
  auto carve = [&](size_t bytes) -> void* {
    void* p = base + off;
    off = (off + bytes + 255) & ~(size_t)255;
    return p;
  };
  float*  h1   = (float*)carve((size_t)NN * NH1 * sizeof(float));
  float*  out1 = (float*)carve((size_t)NN * NH1 * sizeof(float));
  float*  h2   = (float*)carve((size_t)NN * NH2 * sizeof(float));
  float*  out2 = (float*)carve((size_t)NN * NH2 * sizeof(float));
  float*  deg  = (float*)carve((size_t)NN * sizeof(float));
  float*  dinv = (float*)carve((size_t)NN * sizeof(float));
  __bf16* bhi1 = (__bf16*)carve((size_t)KT1 * 2 * 512 * sizeof(__bf16));
  __bf16* blo1 = (__bf16*)carve((size_t)KT1 * 2 * 512 * sizeof(__bf16));
  __bf16* bhi2 = (__bf16*)carve((size_t)512 * sizeof(__bf16));
  __bf16* blo2 = (__bf16*)carve((size_t)512 * sizeof(__bf16));

  const int T = 256;
  auto cdiv = [](long long a, long long b) { return (int)((a + b - 1) / b); };

  // 1) pre-swizzle weights into WMMA-B bf16 hi/lo planes
  prep_bswz<<<cdiv((long long)KT1 * 2 * 512, T), T, 0, stream>>>(W1, NN, NH1, KT1, 2, bhi1, blo1);
  prep_bswz<<<cdiv(512, T), T, 0, stream>>>(W2, NH1, NH2, 1, 1, bhi2, blo2);

  // 2) degrees (self-loop included) and h1 zero-init
  fill_f32<<<cdiv((long long)NN * NH1, T), T, 0, stream>>>(h1, 0.0f, NN * NH1);
  fill_f32<<<cdiv(NN, T), T, 0, stream>>>(deg, 1.0f, NN);
  deg_edges<<<cdiv(E, T), T, 0, stream>>>(edst, deg, E);

  // 3) big GEMM: h1 = x @ W1 (split-K, WMMA bf16 hi/lo, atomic reduce)
  gemm1_wmma<<<cdiv((long long)MT * NCHK * 32, T), T, 0, stream>>>(x, bhi1, blo1, h1);

  // 4) dinv = rsqrt(deg)
  rsqrt_f32<<<cdiv(NN, T), T, 0, stream>>>(deg, dinv, NN);

  // 5) layer-1 aggregation: out1 = b1 + dinv^2*h1 (self) + sum_edges norm*h1[src]
  self_bias<<<cdiv((long long)NN * NH1, T), T, 0, stream>>>(h1, dinv, b1, out1, NN * NH1, 5);
  agg_edges32<<<cdiv((long long)E * 32, T), T, 0, stream>>>(esrc, edst, dinv, h1, out1, E);

  // 6) layer-2 transform: h2 = relu(out1) @ W2 (WMMA, fused ReLU)
  gemm2_wmma<<<cdiv((long long)MT * 32, T), T, 0, stream>>>(out1, bhi2, blo2, h2);

  // 7) layer-2 aggregation
  self_bias<<<cdiv((long long)NN * NH2, T), T, 0, stream>>>(h2, dinv, b2, out2, NN * NH2, 4);
  agg_edges16<<<cdiv((long long)E * 16, T), T, 0, stream>>>(esrc, edst, dinv, h2, out2, E);

  // 8) pair scores
  pair_score<<<cdiv(P, T), T, 0, stream>>>(out2, pi, Wfc, bfc, out, P);
}
